// BGNN4VD_23502061043724
// MI455X (gfx1250) — compile-verified, matrix operand order checked by reading, not run on hardware
//
#include <hip/hip_runtime.h>
#include <hip/hip_bf16.h>
#include <string.h>
#include <stdint.h>

// ---------------- constants from the reference ----------------
#define N_NODES   100000
#define N_EDGES   800000
#define N_GRAPHS  64
#define F_NODE    128
#define F_EDGE    64
#define HID       256
#define HALF      128
#define N_LAYERS  6

typedef __attribute__((ext_vector_type(16))) __bf16          v16bf;
typedef __attribute__((ext_vector_type(16))) unsigned short  v16u;
typedef __attribute__((ext_vector_type(8)))  float           v8f;

// ---------------- device helpers ----------------
__device__ __forceinline__ unsigned dev_f2bf_pack(float f0, float f1) {
  union { float f; unsigned u; } x0, x1;
  x0.f = f0; x1.f = f1;
  unsigned r0 = x0.u + 0x7FFFu + ((x0.u >> 16) & 1u);   // RNE
  unsigned r1 = x1.u + 0x7FFFu + ((x1.u >> 16) & 1u);
  return (r0 >> 16) | (r1 & 0xFFFF0000u);
}
// order-preserving float<->uint transform so atomicMax(uint) == float max
__device__ __forceinline__ unsigned dev_enc_max(float f) {
  union { float f; unsigned u; } x; x.f = f;
  return (x.u & 0x80000000u) ? ~x.u : (x.u | 0x80000000u);
}
__device__ __forceinline__ float dev_dec_max(unsigned v) {
  unsigned u = (v & 0x80000000u) ? (v ^ 0x80000000u) : ~v;
  union { unsigned u; float f; } x; x.u = u;
  return x.f;
}

// ---------------- WMMA bf16 GEMM: C = act(A[MxK] @ B[KxN] + bias) ----------------
// block = 256 threads (8 wave32), block tile 128(M) x 64(N), K step 32.
// Each wave computes a 32x32 output tile: 4 v_wmma_f32_16x16x32_bf16 per K-step.
// LDS panels are stored FRAGMENT-MAJOR (the CDNA5 per-lane K swizzle is applied
// at staging time), so each lane fetches a whole fragment with wide ds_load_b128.
// Requires: K % 32 == 0, N % 64 == 0. M guarded.
__global__ __launch_bounds__(256)
void gemm_wmma_bf16(const float* __restrict__ A, const float* __restrict__ B,
                    const float* __restrict__ bias, float* __restrict__ C,
                    int M, int N, int K, int act)
{
  // AsF[tile][lane] = 16 bf16 fragment elements for A sub-tile `tile` (8 tiles of 16 rows)
  __shared__ v16u AsF[8 * 32];
  __shared__ v16u BsF[4 * 32];
  unsigned* AsW = (unsigned*)AsF;     // word-granular writers
  unsigned* BsW = (unsigned*)BsF;

  const int tid  = threadIdx.x;
  const int wave = tid >> 5;
  const int lane = tid & 31;
  const int half = lane >> 4;
  const int l15  = lane & 15;
  const int mT0  = (wave & 3) * 2;    // 2 consecutive 16-row A sub-tiles
  const int nT0  = (wave >> 2) * 2;   // 2 consecutive 16-col B sub-tiles
  const int blockM = blockIdx.y * 128;
  const int blockN = blockIdx.x * 64;

  v8f acc00 = {}, acc01 = {}, acc10 = {}, acc11 = {};

  for (int k0 = 0; k0 < K; k0 += 32) {
    // ---- stage A panel 128x32 as bf16 fragments (2048 bf16-pairs, 8/thread) ----
#pragma unroll
    for (int t = 0; t < 8; ++t) {
      int j = tid + 256 * t;          // pair index
      int r  = j >> 4;                // 0..127
      int k  = (j & 15) * 2;          // even k
      int gm = blockM + r;
      float f0 = 0.0f, f1 = 0.0f;
      if (gm < M) {
        const float* ap = A + (size_t)gm * K + k0 + k;
        f0 = ap[0]; f1 = ap[1];
      }
      int hk   = (k >> 3) & 1;
      int e    = (k & 7) + ((k & 16) ? 8 : 0);
      int lt   = (r & 15) + 16 * hk;            // target lane
      int tile = r >> 4;
      AsW[((tile * 32 + lt) << 3) + (e >> 1)] = dev_f2bf_pack(f0, f1);
    }
    // ---- stage B panel 32x64 as bf16 fragments (1024 pairs, 4/thread) ----
#pragma unroll
    for (int t = 0; t < 4; ++t) {
      int j = tid + 256 * t;
      int c = j & 63;
      int k = (j >> 6) * 2;
      const float* bp = B + (size_t)(k0 + k) * N + blockN + c;
      float f0 = bp[0];
      float f1 = bp[N];
      int hk   = (k >> 3) & 1;
      int e    = (k & 7) + ((k & 16) ? 8 : 0);
      int lt   = (c & 15) + 16 * hk;
      int tile = c >> 4;
      BsW[((tile * 32 + lt) << 3) + (e >> 1)] = dev_f2bf_pack(f0, f1);
    }
    // prefetch next K panel (global_prefetch_b8)
    if (k0 + 32 < K) {
      int gm = blockM + (tid >> 1);
      if (gm < M) __builtin_prefetch(&A[(size_t)gm * K + k0 + 32 + ((tid & 1) << 4)], 0, 0);
      __builtin_prefetch(&B[(size_t)(k0 + 32 + (tid >> 6)) * N + blockN + (tid & 63)], 0, 0);
    }
    __syncthreads();

    // ---- wide fragment loads: each is 32B -> 2x ds_load_b128 ----
    v16bf a0 = __builtin_bit_cast(v16bf, AsF[(mT0 + 0) * 32 + lane]);
    v16bf a1 = __builtin_bit_cast(v16bf, AsF[(mT0 + 1) * 32 + lane]);
    v16bf b0 = __builtin_bit_cast(v16bf, BsF[(nT0 + 0) * 32 + lane]);
    v16bf b1 = __builtin_bit_cast(v16bf, BsF[(nT0 + 1) * 32 + lane]);

    acc00 = __builtin_amdgcn_wmma_f32_16x16x32_bf16(false, a0, false, b0, (short)0, acc00, false, false);
    acc01 = __builtin_amdgcn_wmma_f32_16x16x32_bf16(false, a0, false, b1, (short)0, acc01, false, false);
    acc10 = __builtin_amdgcn_wmma_f32_16x16x32_bf16(false, a1, false, b0, (short)0, acc10, false, false);
    acc11 = __builtin_amdgcn_wmma_f32_16x16x32_bf16(false, a1, false, b1, (short)0, acc11, false, false);
    __syncthreads();
  }

  // ---- epilogue: C/D layout -> m = v + 8*(lane>>4), n = lane&15 ----
#pragma unroll
  for (int ti = 0; ti < 2; ++ti) {
#pragma unroll
    for (int tj = 0; tj < 2; ++tj) {
      const v8f* acc = (ti == 0) ? (tj == 0 ? &acc00 : &acc01)
                                 : (tj == 0 ? &acc10 : &acc11);
      int nBase = blockN + (nT0 + tj) * 16 + l15;
      float bv = bias ? bias[nBase] : 0.0f;
#pragma unroll
      for (int v = 0; v < 8; ++v) {
        int m = blockM + (mT0 + ti) * 16 + v + 8 * half;
        if (m < M) {
          float xv = (*acc)[v] + bv;
          if (act) xv = fmaxf(xv, 0.0f);
          C[(size_t)m * N + nBase] = xv;
        }
      }
    }
  }
}

// ---------------- fills ----------------
__global__ void fill_f32(float* p, int n, float v) {
  int i = blockIdx.x * blockDim.x + threadIdx.x;
  if (i < n) p[i] = v;
}
__global__ void fill_u32(unsigned* p, int n, unsigned v) {
  int i = blockIdx.x * blockDim.x + threadIdx.x;
  if (i < n) p[i] = v;
}

// ---------------- GAT pieces ----------------
__global__ void mask_kernel(const float* __restrict__ ea, unsigned char* __restrict__ mask, int nE) {
  int e = blockIdx.x * blockDim.x + threadIdx.x;
  if (e < nE) {
    const float* r = ea + (size_t)e * F_EDGE;
    mask[e] = (r[0] == 1.0f) || (r[2] == 1.0f) || (r[4] == 1.0f);
  }
}

// one wave per node: a_src[n] = h[n].att_src, a_dst[n] = h[n].att_dst
__global__ __launch_bounds__(256)
void node_dots(const float* __restrict__ hW, const float* __restrict__ att_src,
               const float* __restrict__ att_dst, float* __restrict__ a_src,
               float* __restrict__ a_dst, int nNodes)
{
  int node = blockIdx.x * 8 + (threadIdx.x >> 5);
  int lane = threadIdx.x & 31;
  if (node >= nNodes) return;
  const float* row = hW + (size_t)node * HALF;
  float s = 0.0f, d = 0.0f;
  for (int c = lane; c < HALF; c += 32) { float h = row[c]; s += h * att_src[c]; d += h * att_dst[c]; }
  for (int o = 16; o; o >>= 1) { s += __shfl_xor(s, o, 32); d += __shfl_xor(d, o, 32); }
  if (lane == 0) { a_src[node] = s; a_dst[node] = d; }
}

// ve = W_edge @ att_edge  (64-vector)
__global__ void ve_kernel(const float* __restrict__ W_edge, const float* __restrict__ att_edge,
                          float* __restrict__ ve)
{
  int i = threadIdx.x;
  if (i < F_EDGE) {
    float s = 0.0f;
    for (int j = 0; j < HALF; ++j) s += W_edge[(size_t)i * HALF + j] * att_edge[j];
    ve[i] = s;
  }
}

// wave per edge: logit = leaky_relu(a_src[src]+a_dst[dst]+ea.ve); mask; segment max.
// ve (64 floats) is staged into LDS once per block via an async LDS load
// (ASYNCcnt-tracked, CDNA5 global_load_async_to_lds_b128).
__global__ __launch_bounds__(256)
void edge_logit_max(const float* __restrict__ ea, const float* __restrict__ ve,
                    const float* __restrict__ a_src, const float* __restrict__ a_dst,
                    const int* __restrict__ src, const int* __restrict__ dst,
                    const unsigned char* __restrict__ mask, int want,
                    float* __restrict__ logit, unsigned* __restrict__ mEnc, int nE)
{
  __shared__ float lve[F_EDGE];   // only LDS object in this kernel -> offset 0
  if (threadIdx.x < 16) {
    unsigned ldsoff = threadIdx.x * 16;          // 16 bytes per lane
    const float* gp = ve + threadIdx.x * 4;
    asm volatile("global_load_async_to_lds_b128 %0, %1, off"
                 :: "v"(ldsoff), "v"(gp) : "memory");
  }
  asm volatile("s_wait_asynccnt 0x0" ::: "memory");
  __syncthreads();

  int e = blockIdx.x * 8 + (threadIdx.x >> 5);
  int lane = threadIdx.x & 31;
  if (e >= nE) return;
  const float* r = ea + (size_t)e * F_EDGE;
  float s = 0.0f;
  for (int c = lane; c < F_EDGE; c += 32) s += r[c] * lve[c];
  for (int o = 16; o; o >>= 1) s += __shfl_xor(s, o, 32);
  if (lane == 0) {
    float x = a_src[src[e]] + a_dst[dst[e]] + s;
    x = (x >= 0.0f) ? x : 0.2f * x;              // leaky_relu 0.2
    if ((mask[e] != 0) != (want != 0)) x = -1e30f;
    logit[e] = x;
    atomicMax(mEnc + dst[e], dev_enc_max(x));
  }
}

__global__ void edge_exp_den(const float* __restrict__ logit, const unsigned* __restrict__ mEnc,
                             const int* __restrict__ dst, const unsigned char* __restrict__ mask,
                             int want, float* __restrict__ ex, float* __restrict__ den, int nE)
{
  int e = blockIdx.x * blockDim.x + threadIdx.x;
  if (e >= nE) return;
  float v = 0.0f;
  if ((mask[e] != 0) == (want != 0)) {
    float m = dev_dec_max(mEnc[dst[e]]);
    v = expf(logit[e] - m);
  }
  ex[e] = v;
  if (v != 0.0f) atomicAdd(den + dst[e], v);
}

// wave per edge: fb[dst, off:off+128] += alpha * hW[src]
__global__ __launch_bounds__(256)
void edge_scatter(const float* __restrict__ ex, const float* __restrict__ den,
                  const float* __restrict__ hW, const int* __restrict__ src,
                  const int* __restrict__ dst, float* __restrict__ fb, int off, int nE)
{
  int e = blockIdx.x * 8 + (threadIdx.x >> 5);
  int lane = threadIdx.x & 31;
  if (e >= nE) return;
  float v = ex[e];
  if (v == 0.0f) return;
  int d = dst[e];
  float a = v / fmaxf(den[d], 1e-16f);
  const float* hs = hW + (size_t)src[e] * HALF;
  float* out = fb + (size_t)d * HID + off;
  for (int c = lane; c < HALF; c += 32) atomicAdd(out + c, a * hs[c]);
}

__global__ void add_bias_half(float* __restrict__ fb, const float* __restrict__ bias,
                              int off, int nNodes)
{
  int idx = blockIdx.x * blockDim.x + threadIdx.x;
  if (idx >= nNodes * HALF) return;
  int n = idx >> 7, c = idx & 127;
  fb[(size_t)n * HID + off + c] += bias[c];
}

// ---------------- batchnorm over axis 0 (256 features) ----------------
__global__ __launch_bounds__(256)
void bn_reduce256(const float* __restrict__ z, float* __restrict__ sum,
                  float* __restrict__ sumsq, int rows)
{
  int col = threadIdx.x;
  int r0 = blockIdx.x * 256;
  int r1 = r0 + 256; if (r1 > rows) r1 = rows;
  float s = 0.0f, q = 0.0f;
  for (int r = r0; r < r1; ++r) { float v = z[(size_t)r * 256 + col]; s += v; q += v * v; }
  atomicAdd(sum + col, s);
  atomicAdd(sumsq + col, q);
}
__global__ void bn_final(float* sum, float* sumsq, float invCnt, int C) {
  int c = blockIdx.x * blockDim.x + threadIdx.x;
  if (c < C) {
    float mu = sum[c] * invCnt;
    float var = sumsq[c] * invCnt - mu * mu;
    sum[c] = mu;
    sumsq[c] = rsqrtf(var + 1e-5f);
  }
}
__global__ void bn_apply_res(const float* __restrict__ z, const float* __restrict__ mu,
                             const float* __restrict__ inv, const float* __restrict__ g,
                             const float* __restrict__ b, float* __restrict__ h, int total)
{
  int i = blockIdx.x * blockDim.x + threadIdx.x;
  if (i >= total) return;
  int c = i & 255;
  float v = (z[i] - mu[c]) * inv[c] * g[c] + b[c];
  h[i] = fmaxf(v, 0.0f) + h[i];
}

// ---------------- pooling ----------------
__global__ void pool_count(const int* __restrict__ batch, float* __restrict__ cnt, int nNodes) {
  int n = blockIdx.x * blockDim.x + threadIdx.x;
  if (n < nNodes) atomicAdd(cnt + batch[n], 1.0f);
}
__global__ void pool_scatter(const float* __restrict__ h, const int* __restrict__ batch,
                             float* __restrict__ msum, unsigned* __restrict__ mx, int total)
{
  int i = blockIdx.x * blockDim.x + threadIdx.x;
  if (i >= total) return;
  int n = i >> 8, f = i & 255;
  int g = batch[n];
  float v = h[i];
  atomicAdd(msum + g * 256 + f, v);
  atomicMax(mx + g * 256 + f, dev_enc_max(v));
}
__global__ void pool_finalize(const float* __restrict__ msum, const unsigned* __restrict__ mx,
                              const float* __restrict__ cnt, float* __restrict__ pooled, int total)
{
  int i = blockIdx.x * blockDim.x + threadIdx.x;
  if (i >= total) return;
  int g = i >> 8, f = i & 255;
  float c = fmaxf(cnt[g], 1.0f);
  float mean = msum[i] / c;
  float m = dev_dec_max(mx[i]);
  if (!(m >= -3.4e38f && m <= 3.4e38f)) m = 0.0f;   // isfinite -> else 0
  pooled[(size_t)g * 512 + f] = mean;
  pooled[(size_t)g * 512 + 256 + f] = m;
}

// ---------------- CNN + FC head ----------------
__global__ void conv1d_k3(const float* __restrict__ in, const float* __restrict__ W,
                          const float* __restrict__ bias, float* __restrict__ out,
                          int B, int Cin, int Cout, int L)
{
  int i = blockIdx.x * blockDim.x + threadIdx.x;
  int total = B * Cout * L;
  if (i >= total) return;
  int p = i % L; int t = i / L; int co = t % Cout; int b = t / Cout;
  float s = bias[co];
  const float* ib = in + (size_t)b * Cin * L;
  const float* wr = W + (size_t)co * Cin * 3;
  for (int ci = 0; ci < Cin; ++ci) {
    const float* ir = ib + (size_t)ci * L;
    const float* w = wr + ci * 3;
    if (p > 0)     s += ir[p - 1] * w[0];
    s += ir[p] * w[1];
    if (p + 1 < L) s += ir[p + 1] * w[2];
  }
  out[i] = s;
}
__global__ void bn_reduce_ch(const float* __restrict__ z, float* __restrict__ sum,
                             float* __restrict__ sumsq, int B, int C, int L)
{
  int i = blockIdx.x * blockDim.x + threadIdx.x;
  int total = B * C * L;
  if (i >= total) return;
  int c = (i / L) % C;
  float v = z[i];
  atomicAdd(sum + c, v);
  atomicAdd(sumsq + c, v * v);
}
__global__ void bn_relu_pool(const float* __restrict__ z, const float* __restrict__ mu,
                             const float* __restrict__ inv, const float* __restrict__ g,
                             const float* __restrict__ beta, float* __restrict__ out,
                             int B, int C, int L)
{
  int Lo = L / 2;
  int i = blockIdx.x * blockDim.x + threadIdx.x;
  int total = B * C * Lo;
  if (i >= total) return;
  int p = i % Lo; int t = i / Lo; int c = t % C; int b = t / C;
  const float* zr = z + ((size_t)b * C + c) * L;
  float s = inv[c] * g[c], m = mu[c], bb = beta[c];
  float y0 = fmaxf((zr[2 * p] - m) * s + bb, 0.0f);
  float y1 = fmaxf((zr[2 * p + 1] - m) * s + bb, 0.0f);
  out[i] = fmaxf(y0, y1);
}
__global__ void fc_kernel(const float* __restrict__ in, const float* __restrict__ W,
                          const float* __restrict__ bias, float* __restrict__ out,
                          int M, int K, int N, int act)
{
  int i = blockIdx.x * blockDim.x + threadIdx.x;
  if (i >= M * N) return;
  int n = i % N, m = i / N;
  float s = bias[n];
  const float* ir = in + (size_t)m * K;
  for (int k = 0; k < K; ++k) s += ir[k] * W[(size_t)k * N + n];
  if (act) s = fmaxf(s, 0.0f);
  out[i] = s;
}

// ---------------- host side ----------------
static unsigned host_enc_max(float f) {
  unsigned u; memcpy(&u, &f, 4);
  return (u & 0x80000000u) ? ~u : (u | 0x80000000u);
}

extern "C" void kernel_launch(void* const* d_in, const int* in_sizes, int n_in,
                              void* d_out, int out_size, void* d_ws, size_t ws_size,
                              hipStream_t stream)
{
  (void)in_sizes; (void)n_in; (void)out_size; (void)ws_size;

  // input layout (setup_inputs dict order, params flattened in insertion order)
  const float* x     = (const float*)d_in[0];
  const int*   ei    = (const int*)d_in[1];
  const float* ea    = (const float*)d_in[2];
  const int*   batch = (const int*)d_in[3];
  const int* src = ei;
  const int* dst = ei + N_EDGES;
  const float* embed_W = (const float*)d_in[4];
  const float* embed_b = (const float*)d_in[5];
  // layers: base 6, 16 tensors each: fwd{W,att_src,att_dst,W_edge,att_edge,bias},
  //         bwd{...}, comb_W, comb_b, bn_g, bn_b
  const float* pool_W = (const float*)d_in[102];
  const float* pool_b = (const float*)d_in[103];
  // convs: 104 + c*4: {W, b, g, beta}; fc: 116 + i*2: {W, b}

  // workspace bump allocator
  char* wp = (char*)d_ws;
  auto alloc = [&](size_t bytes) -> void* {
    void* r = (void*)wp;
    wp += (bytes + 255) & ~(size_t)255;
    return r;
  };
  float* h    = (float*)alloc(sizeof(float) * (size_t)N_NODES * HID);
  float* fb   = (float*)alloc(sizeof(float) * (size_t)N_NODES * HID);
  float* z2   = (float*)alloc(sizeof(float) * (size_t)N_NODES * HID);
  float* hW   = z2;   // overlay: hW is dead before z2 is produced
  float* a_src = (float*)alloc(sizeof(float) * N_NODES);
  float* a_dst = (float*)alloc(sizeof(float) * N_NODES);
  float* ve    = (float*)alloc(sizeof(float) * F_EDGE);
  float* logit = (float*)alloc(sizeof(float) * N_EDGES);
  float* ex    = (float*)alloc(sizeof(float) * N_EDGES);
  unsigned* mEnc = (unsigned*)alloc(sizeof(unsigned) * N_NODES);
  float* den   = (float*)alloc(sizeof(float) * N_NODES);
  unsigned char* emask = (unsigned char*)alloc(N_EDGES);
  float* bnsum = (float*)alloc(sizeof(float) * 256);
  float* bnsq  = (float*)alloc(sizeof(float) * 256);
  float* cnt   = (float*)alloc(sizeof(float) * N_GRAPHS);
  float* psum  = (float*)alloc(sizeof(float) * N_GRAPHS * 256);
  unsigned* pmax = (unsigned*)alloc(sizeof(unsigned) * N_GRAPHS * 256);
  float* pooled = (float*)alloc(sizeof(float) * N_GRAPHS * 512);
  float* gbuf   = (float*)alloc(sizeof(float) * N_GRAPHS * 256);
  float* cbufA  = (float*)alloc(sizeof(float) * (size_t)N_GRAPHS * 128 * 256);
  float* cbufB  = (float*)alloc(sizeof(float) * (size_t)N_GRAPHS * 128 * 128);
  float* fcb1   = (float*)alloc(sizeof(float) * N_GRAPHS * 128);
  float* fcb2   = (float*)alloc(sizeof(float) * N_GRAPHS * 64);

  auto gemm = [&](const float* A, const float* B, const float* bias, float* C,
                  int M, int N, int K, int act) {
    dim3 grid(N / 64, (M + 127) / 128);
    gemm_wmma_bf16<<<grid, 256, 0, stream>>>(A, B, bias, C, M, N, K, act);
  };
  auto fillf = [&](float* p, int n, float v) {
    fill_f32<<<(n + 255) / 256, 256, 0, stream>>>(p, n, v);
  };
  auto fillu = [&](unsigned* p, int n, unsigned v) {
    fill_u32<<<(n + 255) / 256, 256, 0, stream>>>(p, n, v);
  };

  const unsigned encNeg1e30 = host_enc_max(-1e30f);
  const unsigned encNegInf  = host_enc_max(-__builtin_inff());

  mask_kernel<<<(N_EDGES + 255) / 256, 256, 0, stream>>>(ea, emask, N_EDGES);

  // embed: h = relu(x @ embed_W + embed_b)
  gemm(x, embed_W, embed_b, h, N_NODES, HID, F_NODE, 1);

  for (int L = 0; L < N_LAYERS; ++L) {
    int base = 6 + 16 * L;
    fillf(fb, N_NODES * HID, 0.0f);

    for (int dir = 0; dir < 2; ++dir) {
      int pofs = base + dir * 6;
      const float* W        = (const float*)d_in[pofs + 0];
      const float* att_s    = (const float*)d_in[pofs + 1];
      const float* att_d    = (const float*)d_in[pofs + 2];
      const float* W_edge   = (const float*)d_in[pofs + 3];
      const float* att_edge = (const float*)d_in[pofs + 4];
      const float* bias     = (const float*)d_in[pofs + 5];
      int want = (dir == 0) ? 1 : 0;
      int off  = (dir == 0) ? 0 : HALF;

      gemm(h, W, nullptr, hW, N_NODES, HALF, HID, 0);
      node_dots<<<(N_NODES + 7) / 8, 256, 0, stream>>>(hW, att_s, att_d, a_src, a_dst, N_NODES);
      ve_kernel<<<1, 64, 0, stream>>>(W_edge, att_edge, ve);
      fillu(mEnc, N_NODES, encNeg1e30);
      fillf(den, N_NODES, 0.0f);
      edge_logit_max<<<(N_EDGES + 7) / 8, 256, 0, stream>>>(
          ea, ve, a_src, a_dst, src, dst, emask, want, logit, mEnc, N_EDGES);
      edge_exp_den<<<(N_EDGES + 255) / 256, 256, 0, stream>>>(
          logit, mEnc, dst, emask, want, ex, den, N_EDGES);
      edge_scatter<<<(N_EDGES + 7) / 8, 256, 0, stream>>>(
          ex, den, hW, src, dst, fb, off, N_EDGES);
      add_bias_half<<<(N_NODES * HALF + 255) / 256, 256, 0, stream>>>(fb, bias, off, N_NODES);
    }

    const float* comb_W = (const float*)d_in[base + 12];
    const float* comb_b = (const float*)d_in[base + 13];
    const float* bn_g   = (const float*)d_in[base + 14];
    const float* bn_b   = (const float*)d_in[base + 15];

    gemm(fb, comb_W, comb_b, z2, N_NODES, HID, HID, 0);
    fillf(bnsum, 256, 0.0f);
    fillf(bnsq, 256, 0.0f);
    bn_reduce256<<<(N_NODES + 255) / 256, 256, 0, stream>>>(z2, bnsum, bnsq, N_NODES);
    bn_final<<<1, 256, 0, stream>>>(bnsum, bnsq, 1.0f / (float)N_NODES, 256);
    bn_apply_res<<<(N_NODES * HID + 255) / 256, 256, 0, stream>>>(
        z2, bnsum, bnsq, bn_g, bn_b, h, N_NODES * HID);
  }

  // ---- graph pooling ----
  fillf(cnt, N_GRAPHS, 0.0f);
  fillf(psum, N_GRAPHS * 256, 0.0f);
  fillu(pmax, N_GRAPHS * 256, encNegInf);
  pool_count<<<(N_NODES + 255) / 256, 256, 0, stream>>>(batch, cnt, N_NODES);
  pool_scatter<<<(N_NODES * HID + 255) / 256, 256, 0, stream>>>(h, batch, psum, pmax, N_NODES * HID);
  pool_finalize<<<(N_GRAPHS * 256 + 255) / 256, 256, 0, stream>>>(psum, pmax, cnt, pooled, N_GRAPHS * 256);
  gemm(pooled, pool_W, pool_b, gbuf, N_GRAPHS, HID, 512, 1);   // relu

  // ---- CNN head: (Cin,L): (1,256)->(128,128)->(64,64)->(32,32) ----
  const int chans[4] = {1, 128, 64, 32};
  const int lens[4]  = {256, 256, 128, 64};
  const float* cin = gbuf;
  for (int s = 0; s < 3; ++s) {
    const float* cW = (const float*)d_in[104 + s * 4 + 0];
    const float* cb = (const float*)d_in[104 + s * 4 + 1];
    const float* cg = (const float*)d_in[104 + s * 4 + 2];
    const float* cbeta = (const float*)d_in[104 + s * 4 + 3];
    int Cin = chans[s], Cout = chans[s + 1], L = lens[s + 1];
    int totC = N_GRAPHS * Cout * L;
    conv1d_k3<<<(totC + 255) / 256, 256, 0, stream>>>(cin, cW, cb, cbufA, N_GRAPHS, Cin, Cout, L);
    fillf(bnsum, 256, 0.0f);
    fillf(bnsq, 256, 0.0f);
    bn_reduce_ch<<<(totC + 255) / 256, 256, 0, stream>>>(cbufA, bnsum, bnsq, N_GRAPHS, Cout, L);
    bn_final<<<1, 256, 0, stream>>>(bnsum, bnsq, 1.0f / (float)(N_GRAPHS * L), Cout);
    int totP = N_GRAPHS * Cout * (L / 2);
    bn_relu_pool<<<(totP + 255) / 256, 256, 0, stream>>>(cbufA, bnsum, bnsq, cg, cbeta, cbufB, N_GRAPHS, Cout, L);
    cin = cbufB;
  }

  // ---- FC head: 1024 -> 128 -> 64 -> 2 ----
  const float* fcW0 = (const float*)d_in[116]; const float* fcb0 = (const float*)d_in[117];
  const float* fcW1 = (const float*)d_in[118]; const float* fcb1_ = (const float*)d_in[119];
  const float* fcW2 = (const float*)d_in[120]; const float* fcb2_ = (const float*)d_in[121];
  fc_kernel<<<(N_GRAPHS * 128 + 255) / 256, 256, 0, stream>>>(cbufB, fcW0, fcb0, fcb1, N_GRAPHS, 1024, 128, 1);
  fc_kernel<<<(N_GRAPHS * 64 + 255) / 256, 256, 0, stream>>>(fcb1, fcW1, fcb1_, fcb2, N_GRAPHS, 128, 64, 1);
  fc_kernel<<<(N_GRAPHS * 2 + 255) / 256, 256, 0, stream>>>(fcb2, fcW2, fcb2_, (float*)d_out, N_GRAPHS, 64, 2, 0);
}
